// GCN_73770358276469
// MI455X (gfx1250) — matrix operand order repaired and checked by reading, not compiled, not run on hardware
//
#include <hip/hip_runtime.h>

// ---------------------------------------------------------------------------
// GCN (Chebyshev graph conv) for MI455X / gfx1250.
// Heavy GEMMs: v_wmma_f32_16x16x32_bf16, all operands via contiguous 128-bit
// vector loads (B operands pre-transposed so K is the fastest axis).
// ---------------------------------------------------------------------------

typedef __bf16 bf16_t;
typedef bf16_t v16bf __attribute__((ext_vector_type(16)));
typedef bf16_t v8bf  __attribute__((ext_vector_type(8)));
typedef float  v8f   __attribute__((ext_vector_type(8)));

#define B_    64
#define CIN   32
#define NN    1024
#define LL    12
#define KORD  3
#define EMB   10
#define COUT  64
#define BL_   (B_ * LL)          // 768
#define NCOL  (B_ * CIN * LL)    // 24576 GEMM columns

// ---------------- x transpose: xT[(bc*LL+l)*NN + m] = bf16(x[bc, m, l]) -----
__global__ void k_transpose_x(const float* __restrict__ x,
                              bf16_t* __restrict__ xT) {
  const int total = NCOL * NN;
  int o = blockIdx.x * blockDim.x + threadIdx.x;
  int stride = gridDim.x * blockDim.x;
  for (; o < total; o += stride) {
    int m = o & (NN - 1);
    int j = o >> 10;                 // j = bc*LL + l
    int bc = j / LL, l = j - bc * LL;
    xT[o] = (bf16_t)x[((size_t)bc * NN + m) * LL + l];
  }
}

// -------- a -> bf16 (row-major) + transposed bf16 ---------------------------
__global__ void k_conv_a(const float* __restrict__ a,
                         bf16_t* __restrict__ abf,
                         bf16_t* __restrict__ aT) {
  const int total = NN * NN;
  int o = blockIdx.x * blockDim.x + threadIdx.x;
  int stride = gridDim.x * blockDim.x;
  for (; o < total; o += stride) {
    int n = o >> 10, m = o & (NN - 1);
    bf16_t v = (bf16_t)a[o];
    abf[o] = v;
    aT[(size_t)m * NN + n] = v;
  }
}

// ---------------- node_emb[n,d] = a0[n,:] . W_ne[d,:] + b_ne[d] -------------
__global__ void k_node_emb(const float* __restrict__ a0,
                           const float* __restrict__ Wne,
                           const float* __restrict__ bne,
                           float* __restrict__ ne) {
  int n = blockIdx.x;
  int lane = threadIdx.x;                     // 32 threads = 1 wave
  const float* arow = a0 + (size_t)n * NN;
  for (int d = 0; d < EMB; ++d) {
    const float* wrow = Wne + (size_t)d * NN;
    float s = 0.f;
    for (int m = lane; m < NN; m += 32) s += arow[m] * wrow[m];
    for (int off = 16; off > 0; off >>= 1) s += __shfl_xor(s, off, 32);
    if (lane == 0) ne[n * EMB + d] = s + bne[d];
  }
}

// ---------------- Wn[n, k*1024 + i*32 + o] = sum_d ne[n,d] * wp[d,...] ------
__global__ void k_node_weights(const float* __restrict__ ne,
                               const float* __restrict__ wp,
                               float* __restrict__ Wn) {
  const int per = KORD * CIN * CIN;           // 3072
  const int total = NN * per;
  int idx = blockIdx.x * blockDim.x + threadIdx.x;
  int stride = gridDim.x * blockDim.x;
  for (; idx < total; idx += stride) {
    int n = idx / per;
    int r = idx - n * per;
    float s = 0.f;
#pragma unroll
    for (int d = 0; d < EMB; ++d) s += ne[n * EMB + d] * wp[(size_t)d * per + r];
    Wn[idx] = s;
  }
}

__global__ void k_node_bias(const float* __restrict__ ne,
                            const float* __restrict__ bp,
                            float* __restrict__ biasn) {
  const int total = NN * CIN;
  int idx = blockIdx.x * blockDim.x + threadIdx.x;
  int stride = gridDim.x * blockDim.x;
  for (; idx < total; idx += stride) {
    int n = idx / CIN, c = idx - n * CIN;
    float s = 0.f;
#pragma unroll
    for (int d = 0; d < EMB; ++d) s += ne[n * EMB + d] * bp[d * CIN + c];
    biasn[idx] = s;
  }
}

// -------- A fragment: A[m,k] = S[row0+m, kk+k]; row-major bf16, ld=NN -------
// lane: m = lane&15, hi = lane>>4; element e -> K=hi*16+e
__device__ __forceinline__ v16bf load_a_frag(const bf16_t* __restrict__ S,
                                             int row0, int kk, int m, int hi) {
  const bf16_t* ar = S + (size_t)(row0 + m) * NN + kk;
  v8bf lo = *(const v8bf*)(ar + hi * 16);
  v8bf hv = *(const v8bf*)(ar + hi * 16 + 8);
  return __builtin_shufflevector(lo, hv, 0, 1, 2, 3, 4, 5, 6, 7,
                                 8, 9, 10, 11, 12, 13, 14, 15);
}

// -------- B fragment from K-contiguous (transposed) storage: ----------------
// B[k,j] = T[col(j)*NN + kk + k]; lane j = col0+m, elements K = hi*16 + 0..15
__device__ __forceinline__ v16bf load_b_frag(const bf16_t* __restrict__ T,
                                             int col, int kk, int hi) {
  return *(const v16bf*)(T + (size_t)col * NN + kk + hi * 16);
}

// ---------------- T2 = 2*a@a - I  (bf16 in/out, f32 accumulate) -------------
__global__ void k_t2_gemm(const bf16_t* __restrict__ abf,
                          const bf16_t* __restrict__ aT,
                          bf16_t* __restrict__ t2) {
  int lane = threadIdx.x & 31;
  int m = lane & 15, hi = lane >> 4;
  int row0 = blockIdx.y * 16;
  int col0 = blockIdx.x * 16;
  v8f acc = {};
  for (int kk = 0; kk < NN; kk += 32) {
    v16bf Av = load_a_frag(abf, row0, kk, m, hi);
    v16bf Bv = load_b_frag(aT, col0 + m, kk, hi);
    acc = __builtin_amdgcn_wmma_f32_16x16x32_bf16(false, Av, false, Bv,
                                                  (short)0, acc, false, false);
  }
#pragma unroll
  for (int r = 0; r < 8; ++r) {
    int R = row0 + r + 8 * hi;
    int Cc = col0 + m;
    float v = 2.f * acc[r] - (R == Cc ? 1.f : 0.f);
    t2[(size_t)R * NN + Cc] = (bf16_t)v;
  }
}

// ------- x_g[b,kidx,n,c,l] = sum_m S[n,m] * x[b,c,m,l] ----------------------
// 32x64 output per wave: 2 A-frags x 4 B-frags -> 8 WMMA per 32-K chunk.
__global__ void k_support_gemm(const bf16_t* __restrict__ S,
                               const bf16_t* __restrict__ xT,
                               bf16_t* __restrict__ xg, int kidx) {
  int lane = threadIdx.x & 31;
  int m = lane & 15, hi = lane >> 4;
  int row0 = blockIdx.y * 32;                 // n tile (32 rows)
  int col0 = blockIdx.x * 64;                 // 4 column tiles of 16
  v8f acc[2][4] = {};
  for (int kk = 0; kk < NN; kk += 32) {
    v16bf A0 = load_a_frag(S, row0, kk, m, hi);
    v16bf A1 = load_a_frag(S, row0 + 16, kk, m, hi);
    if (kk + 32 < NN) {                       // gfx1250 global_prefetch_b8
      __builtin_prefetch(S + (size_t)(row0 + m) * NN + kk + 32, 0, 1);
      __builtin_prefetch(xT + (size_t)(col0 + m) * NN + kk + 32, 0, 1);
    }
#pragma unroll
    for (int t = 0; t < 4; ++t) {
      v16bf Bv = load_b_frag(xT, col0 + t * 16 + m, kk, hi);
      acc[0][t] = __builtin_amdgcn_wmma_f32_16x16x32_bf16(
          false, A0, false, Bv, (short)0, acc[0][t], false, false);
      acc[1][t] = __builtin_amdgcn_wmma_f32_16x16x32_bf16(
          false, A1, false, Bv, (short)0, acc[1][t], false, false);
    }
  }
#pragma unroll
  for (int i2 = 0; i2 < 2; ++i2) {
#pragma unroll
    for (int t = 0; t < 4; ++t) {
      int j = col0 + t * 16 + m;              // col = (b*CIN+c)*LL + l
      int bc = j / LL, l = j - bc * LL;
      int b = bc / CIN, c = bc - b * CIN;
#pragma unroll
      for (int r = 0; r < 8; ++r) {
        int R = row0 + i2 * 16 + r + 8 * hi;
        size_t o = ((((size_t)b * KORD + kidx) * NN + R) * CIN + c) * LL + l;
        xg[o] = (bf16_t)acc[i2][t][r];
      }
    }
  }
}

// ---------------- k=0 slice: x_g[b,0,n,c,l] = x[b,c,n,l] --------------------
__global__ void k_xg_k0(const float* __restrict__ x,
                        bf16_t* __restrict__ xg) {
  const int total = B_ * NN * CIN * LL;
  int idx = blockIdx.x * blockDim.x + threadIdx.x;
  int stride = gridDim.x * blockDim.x;
  for (; idx < total; idx += stride) {
    int l = idx % LL; int t = idx / LL;
    int c = t % CIN; t /= CIN;
    int n = t % NN;  int b = t / NN;
    xg[(((size_t)b * KORD * NN + n) * CIN + c) * LL + l] =
        (bf16_t)x[(((size_t)b * CIN + c) * NN + n) * LL + l];
  }
}

// -------- gconv: out2[bl,n,o] = bias[n,o] + sum_{k,i} xg_flat * Wn ----------
// (torch reshape/permute is free: reshape indices address the same flat buffer)
__global__ void k_gconv(const bf16_t* __restrict__ xg,
                        const float* __restrict__ Wn,
                        const float* __restrict__ biasn,
                        float* __restrict__ out2) {
  int blk = blockIdx.x;                       // blk = bl*NN + n
  int n = blk & (NN - 1);
  int bl = blk >> 10;
  int o = threadIdx.x;                        // 32 threads
  __shared__ float xs[KORD * CIN];
  const bf16_t* xgp = xg + (size_t)bl * (KORD * NN * CIN) + (size_t)n * CIN;
  for (int j = o; j < KORD * CIN; j += 32)
    xs[j] = (float)xgp[(size_t)(j / CIN) * (NN * CIN) + (j % CIN)];
  __syncthreads();
  float acc = biasn[n * CIN + o];
  const float* w = Wn + (size_t)n * (KORD * CIN * CIN) + o;
#pragma unroll 8
  for (int j = 0; j < KORD * CIN; ++j) acc += xs[j] * w[(size_t)j * CIN];
  out2[(size_t)bl * (NN * CIN) + n * CIN + o] = acc;
}

// -------- final 1x1 conv, accumulated one branch at a time ------------------
// h[b, br*32+c, n, l] == out2_br flat at ((b*32+c)*NN + n)*LL + l
__global__ void k_mlp(const float* __restrict__ out2,
                      const float* __restrict__ Wmlp,
                      const float* __restrict__ bmlp,
                      float* __restrict__ out, int br) {
  const int total = B_ * COUT * NN * LL;      // 50,331,648
  int idx = blockIdx.x * blockDim.x + threadIdx.x;
  int stride = gridDim.x * blockDim.x;
  for (; idx < total; idx += stride) {
    int l = idx % LL; int t = idx / LL;
    int n = t % NN;  t /= NN;
    int oc = t % COUT; int b = t / COUT;
    float acc = (br == 0) ? bmlp[oc] : out[idx];
    const float* w = Wmlp + (size_t)oc * (2 * CIN) + br * CIN;
    const float* h = out2 + ((size_t)b * CIN * NN + n) * LL + l;
#pragma unroll 8
    for (int c = 0; c < CIN; ++c) acc += w[c] * h[(size_t)c * (NN * LL)];
    out[idx] = acc;
  }
}

// ---------------------------------------------------------------------------
extern "C" void kernel_launch(void* const* d_in, const int* in_sizes, int n_in,
                              void* d_out, int out_size, void* d_ws,
                              size_t ws_size, hipStream_t stream) {
  (void)in_sizes; (void)n_in; (void)out_size; (void)ws_size;
  const float* x    = (const float*)d_in[0];
  const float* a0   = (const float*)d_in[1];
  const float* a1   = (const float*)d_in[2];
  const float* Wne  = (const float*)d_in[3];
  const float* bne  = (const float*)d_in[4];
  const float* wp   = (const float*)d_in[5];
  const float* bp   = (const float*)d_in[6];
  const float* Wmlp = (const float*)d_in[7];
  const float* bmlp = (const float*)d_in[8];
  float* out = (float*)d_out;

  // workspace carve-up (~307 MB)
  char* ws = (char*)d_ws;
  bf16_t* xT   = (bf16_t*)ws; ws += (size_t)NCOL * NN * sizeof(bf16_t);
  bf16_t* abf  = (bf16_t*)ws; ws += (size_t)NN * NN * sizeof(bf16_t);
  bf16_t* aT   = (bf16_t*)ws; ws += (size_t)NN * NN * sizeof(bf16_t);
  bf16_t* t2   = (bf16_t*)ws; ws += (size_t)NN * NN * sizeof(bf16_t);
  float* ne    = (float*)ws;  ws += (size_t)NN * EMB * sizeof(float);
  float* Wn    = (float*)ws;  ws += (size_t)NN * KORD * CIN * CIN * sizeof(float);
  float* biasn = (float*)ws;  ws += (size_t)NN * CIN * sizeof(float);
  bf16_t* xg   = (bf16_t*)ws; ws += (size_t)B_ * KORD * NN * CIN * LL * sizeof(bf16_t);
  float* out2  = (float*)ws;  // BL_ * NN * CIN floats

  k_transpose_x<<<4096, 256, 0, stream>>>(x, xT);
  k_node_emb<<<NN, 32, 0, stream>>>(a0, Wne, bne, ne);
  k_node_weights<<<2048, 256, 0, stream>>>(ne, wp, Wn);
  k_node_bias<<<128, 256, 0, stream>>>(ne, bp, biasn);

  for (int br = 0; br < 2; ++br) {
    const float* a = br ? a1 : a0;
    k_conv_a<<<2048, 256, 0, stream>>>(a, abf, aT);
    k_t2_gemm<<<dim3(64, 64), 32, 0, stream>>>(abf, aT, t2);
    k_xg_k0<<<4096, 256, 0, stream>>>(x, xg);
    k_support_gemm<<<dim3(384, 32), 32, 0, stream>>>(abf, xT, xg, 1);
    k_support_gemm<<<dim3(384, 32), 32, 0, stream>>>(t2, xT, xg, 2);
    k_gconv<<<BL_ * NN, 32, 0, stream>>>(xg, Wn, biasn, out2);
    k_mlp<<<8192, 256, 0, stream>>>(out2, Wmlp, bmlp, out, br);
  }
}
